// LoRAMoE_20160576487591
// MI455X (gfx1250) — matrix operand order repaired and checked by reading, not compiled, not run on hardware
//
#include <hip/hip_runtime.h>
#include <hip/hip_bf16.h>

// Problem constants (match reference)
#define BSZ 8
#define SEQ 2048
#define NTOK (BSZ * SEQ)   // 16384
#define DIM 1024
#define OUT 1024
#define NEXP 8
#define RANK 4
#define TOPK 2
#define SCALING 1.0f

// GEMM tiling
#define BM 128
#define BN 64
#define KB 64              // K per staging round (2 WMMA K-steps of 32)
#define NIT (DIM / KB)     // 16 pipeline iterations

typedef __bf16 bf16;
typedef bf16  bf16x8  __attribute__((ext_vector_type(8)));
typedef bf16  bf16x16 __attribute__((ext_vector_type(16)));
typedef float f32x4   __attribute__((ext_vector_type(4)));
typedef float f32x8   __attribute__((ext_vector_type(8)));

__device__ __forceinline__ float wave_reduce_add(float v) {
#pragma unroll
    for (int off = 16; off > 0; off >>= 1)
        v += __shfl_xor(v, off, 32);
    return v;
}

// -------- Kernel 1: gate (softmax top-2) + rank-4 LoRA "t" contraction --------
__global__ __launch_bounds__(256) void gate_lora_t_kernel(
    const float* __restrict__ x,      // [NTOK, DIM]
    const float* __restrict__ gw,     // [NEXP, DIM]
    const float* __restrict__ left,   // [NEXP, DIM, RANK]
    float* __restrict__ coef,         // [NTOK, TOPK, RANK]
    int*   __restrict__ idx)          // [NTOK, TOPK]
{
    const int lane = threadIdx.x & 31;
    const int wave = threadIdx.x >> 5;
    const int tok  = blockIdx.x * 8 + wave;

    float xr[32];
#pragma unroll
    for (int i = 0; i < 32; ++i)
        xr[i] = x[(size_t)tok * DIM + i * 32 + lane];

    float acc[NEXP];
#pragma unroll
    for (int e = 0; e < NEXP; ++e) acc[e] = 0.0f;
    for (int i = 0; i < 32; ++i) {
        const float xv = xr[i];
        const int d = i * 32 + lane;
#pragma unroll
        for (int e = 0; e < NEXP; ++e)
            acc[e] += xv * gw[e * DIM + d];
    }
#pragma unroll
    for (int e = 0; e < NEXP; ++e) acc[e] = wave_reduce_add(acc[e]);

    float mx = acc[0];
#pragma unroll
    for (int e = 1; e < NEXP; ++e) mx = fmaxf(mx, acc[e]);
    float den = 0.0f;
    float sc[NEXP];
#pragma unroll
    for (int e = 0; e < NEXP; ++e) { sc[e] = __expf(acc[e] - mx); den += sc[e]; }
    const float inv = 1.0f / den;
#pragma unroll
    for (int e = 0; e < NEXP; ++e) sc[e] *= inv;

    int e0 = 0;
#pragma unroll
    for (int e = 1; e < NEXP; ++e) if (sc[e] > sc[e0]) e0 = e;
    int e1 = (e0 == 0) ? 1 : 0;
#pragma unroll
    for (int e = 0; e < NEXP; ++e) if (e != e0 && sc[e] > sc[e1]) e1 = e;
    const int   esel[TOPK] = { e0, e1 };
    const float wsel[TOPK] = { sc[e0], sc[e1] };

#pragma unroll
    for (int k = 0; k < TOPK; ++k) {
        const int e = esel[k];
        f32x4 t = { 0.f, 0.f, 0.f, 0.f };
        for (int i = 0; i < 32; ++i) {
            const int d = i * 32 + lane;
            const f32x4 lv = *(const f32x4*)&left[((size_t)e * DIM + d) * RANK];
            t += xr[i] * lv;
        }
        float tr[RANK];
#pragma unroll
        for (int r = 0; r < RANK; ++r) tr[r] = wave_reduce_add(t[r]);
        if (lane == 0) {
            idx[tok * TOPK + k] = e;
#pragma unroll
            for (int r = 0; r < RANK; ++r)
                coef[(tok * TOPK + k) * RANK + r] = wsel[k] * SCALING * tr[r];
        }
    }
}

// -------- Kernel 2: fused bf16-WMMA GEMM (x @ W^T + b) + LoRA epilogue --------
__device__ __forceinline__ f32x8 wmma_bf16(const bf16x16& a, const bf16x16& b, const f32x8& c) {
    return __builtin_amdgcn_wmma_f32_16x16x32_bf16(
        /*neg_a=*/false, a, /*neg_b=*/false, b,
        /*c_mod=*/(short)0, c, /*reuse_a=*/false, /*reuse_b=*/false);
}

// Register-staged tile fragments (f32 global load -> packed bf16 regs)
struct RegA { bf16x8 p[4]; };   // 32 floats/thread -> 4x bf16x8
struct RegB { bf16x8 p[2]; };   // 16 floats/thread -> 2x bf16x8

__device__ __forceinline__ RegA load_a(const float* __restrict__ x,
                                       size_t tileM, int kb, int t) {
    const int row = t >> 1;
    const int kc  = (t & 1) << 5;                          // 0 or 32
    const float* s = x + (tileM + row) * (size_t)DIM + kb + kc;
    RegA r;
#pragma unroll
    for (int j = 0; j < 4; ++j) {
        f32x4 v0 = ((const f32x4*)s)[2 * j];
        f32x4 v1 = ((const f32x4*)s)[2 * j + 1];
#pragma unroll
        for (int i = 0; i < 4; ++i) { r.p[j][i] = (bf16)v0[i]; r.p[j][i + 4] = (bf16)v1[i]; }
    }
    return r;
}
__device__ __forceinline__ void store_a(const RegA& r, bf16* __restrict__ buf, int t) {
    const int row = t >> 1;
    const int kc  = (t & 1) << 5;
    bf16* dst = &buf[row * KB + kc];
#pragma unroll
    for (int j = 0; j < 4; ++j) *(bf16x8*)(dst + 8 * j) = r.p[j];
}

__device__ __forceinline__ RegB load_b(const float* __restrict__ W,
                                       size_t tileN, int kb, int t) {
    const int n  = t >> 2;
    const int kc = (t & 3) << 4;                           // 0,16,32,48
    const float* s = W + (tileN + n) * (size_t)DIM + kb + kc;
    RegB r;
#pragma unroll
    for (int j = 0; j < 2; ++j) {
        f32x4 v0 = ((const f32x4*)s)[2 * j];
        f32x4 v1 = ((const f32x4*)s)[2 * j + 1];
#pragma unroll
        for (int i = 0; i < 4; ++i) { r.p[j][i] = (bf16)v0[i]; r.p[j][i + 4] = (bf16)v1[i]; }
    }
    return r;
}
__device__ __forceinline__ void store_b(const RegB& r, bf16* __restrict__ buf, int t) {
    const int n  = t >> 2;
    const int kc = (t & 3) << 4;
    bf16* dst = &buf[n * KB + kc];
#pragma unroll
    for (int j = 0; j < 2; ++j) *(bf16x8*)(dst + 8 * j) = r.p[j];
}

// One 32-wide K step: load frags per ISA 7.12.2 layout, 4 WMMAs
__device__ __forceinline__ void mma_step(const bf16* __restrict__ bufA,
                                         const bf16* __restrict__ bufB,
                                         int wm, int wn, int l, int half, int ks,
                                         f32x8 c[2][2]) {
    const int ko = ks * 32;
    bf16x16 a0, a1, b0, b1;
    {
        const bf16* base = &bufA[(wm + l) * KB + ko];
        bf16x8 lo = *(const bf16x8*)(base + half * 8);
        bf16x8 hi = *(const bf16x8*)(base + 16 + half * 8);
#pragma unroll
        for (int i = 0; i < 8; ++i) { a0[i] = lo[i]; a0[i + 8] = hi[i]; }
        base = &bufA[(wm + 16 + l) * KB + ko];
        lo = *(const bf16x8*)(base + half * 8);
        hi = *(const bf16x8*)(base + 16 + half * 8);
#pragma unroll
        for (int i = 0; i < 8; ++i) { a1[i] = lo[i]; a1[i + 8] = hi[i]; }
    }
    {
        const bf16* base = &bufB[(wn + l) * KB + ko];
        bf16x8 q0 = *(const bf16x8*)(base + half * 8);
        bf16x8 q1 = *(const bf16x8*)(base + 16 + half * 8);
#pragma unroll
        for (int i = 0; i < 8; ++i) { b0[i] = q0[i]; b0[i + 8] = q1[i]; }
        base = &bufB[(wn + 16 + l) * KB + ko];
        q0 = *(const bf16x8*)(base + half * 8);
        q1 = *(const bf16x8*)(base + 16 + half * 8);
#pragma unroll
        for (int i = 0; i < 8; ++i) { b1[i] = q0[i]; b1[i + 8] = q1[i]; }
    }
    c[0][0] = wmma_bf16(a0, b0, c[0][0]);
    c[0][1] = wmma_bf16(a0, b1, c[0][1]);
    c[1][0] = wmma_bf16(a1, b0, c[1][0]);
    c[1][1] = wmma_bf16(a1, b1, c[1][1]);
}

__global__ __launch_bounds__(256) void lora_moe_gemm_kernel(
    const float* __restrict__ x,      // [NTOK, DIM]
    const float* __restrict__ W,      // [OUT, DIM]
    const float* __restrict__ bias,   // [OUT]
    const float* __restrict__ right,  // [NEXP, RANK, OUT]
    const float* __restrict__ coef,   // [NTOK, TOPK, RANK]
    const int*   __restrict__ idx,    // [NTOK, TOPK]
    float* __restrict__ out)          // [NTOK, OUT]
{
    __shared__ bf16  lA[2][BM * KB];              // 2 x 16 KB
    __shared__ bf16  lB[2][BN * KB];              // 2 x  8 KB
    __shared__ float right_s[NEXP * RANK * BN];   // 8 KB
    __shared__ float coef_s[BM * TOPK * RANK];    // 4 KB
    __shared__ int   idx_s[BM * TOPK];            // 1 KB
    __shared__ float bias_s[BN];

    const int t    = threadIdx.x;
    const int lane = t & 31;
    const int wave = t >> 5;
    const int wm   = (wave & 3) * 32;
    const int wn   = (wave >> 2) * 32;
    const int l    = lane & 15;
    const int half = lane >> 4;
    const size_t tileM = (size_t)blockIdx.x * BM;
    const size_t tileN = (size_t)blockIdx.y * BN;

    // ---- stage epilogue data (visibility covered by first loop barrier) ----
    {
        const int er = t >> 3;
        const int oc = (t & 7) * 8;
        const float* src = right + (size_t)er * OUT + tileN + oc;
        *(f32x4*)&right_s[er * BN + oc]     = ((const f32x4*)src)[0];
        *(f32x4*)&right_s[er * BN + oc + 4] = ((const f32x4*)src)[1];
    }
    if (t < BM) {
        const f32x4* cp = (const f32x4*)&coef[(tileM + t) * TOPK * RANK];
        *(f32x4*)&coef_s[t * 8]     = cp[0];
        *(f32x4*)&coef_s[t * 8 + 4] = cp[1];
        idx_s[t * 2]     = idx[(tileM + t) * 2];
        idx_s[t * 2 + 1] = idx[(tileM + t) * 2 + 1];
    }
    if (t < BN) bias_s[t] = bias[tileN + t];

    f32x8 c[2][2] = { { {0}, {0} }, { {0}, {0} } };

    // ---- software-pipelined, double-buffered K loop ----
    RegA ra = load_a(x, tileM, 0, t);
    RegB rb = load_b(W, tileN, 0, t);
    store_a(ra, lA[0], t);
    store_b(rb, lB[0], t);

    int cur = 0;
    for (int it = 0; it < NIT; ++it) {
        __syncthreads();   // buf[cur] fully written; buf[cur^1] reads from prev iter done
        const bool more = (it + 1 < NIT);
        if (more) {                           // issue next tile's global loads early
            ra = load_a(x, tileM, (it + 1) * KB, t);
            rb = load_b(W, tileN, (it + 1) * KB, t);
        }
        if (it + 2 < NIT) {                   // prefetch tile after next into caches
            __builtin_prefetch(x + (tileM + (t >> 1)) * (size_t)DIM +
                               (it + 2) * KB + ((t & 1) << 5), 0, 1);
        }
        mma_step(lA[cur], lB[cur], wm, wn, l, half, 0, c);
        mma_step(lA[cur], lB[cur], wm, wn, l, half, 1, c);
        if (more) {
            store_a(ra, lA[cur ^ 1], t);
            store_b(rb, lB[cur ^ 1], t);
        }
        cur ^= 1;
    }

    // ---- epilogue: out = C + bias + sum_k sum_r coef[m][k][r]*right[e_k][r][o] ----
#pragma unroll
    for (int am = 0; am < 2; ++am) {
#pragma unroll
        for (int bn = 0; bn < 2; ++bn) {
#pragma unroll
            for (int r = 0; r < 8; ++r) {
                const int m = wm + am * 16 + half * 8 + r;  // local row (0..127)
                const int o = wn + bn * 16 + l;             // local col (0..63)
                float accv = c[am][bn][r] + bias_s[o];
#pragma unroll
                for (int k = 0; k < TOPK; ++k) {
                    const int e = idx_s[m * 2 + k];
                    const float* rp = &right_s[(e * RANK) * BN + o];
                    const float* cp = &coef_s[m * 8 + k * 4];
                    accv += cp[0] * rp[0] + cp[1] * rp[BN] +
                            cp[2] * rp[2 * BN] + cp[3] * rp[3 * BN];
                }
                out[(tileM + m) * (size_t)OUT + tileN + o] = accv;
            }
        }
    }
}

extern "C" void kernel_launch(void* const* d_in, const int* in_sizes, int n_in,
                              void* d_out, int out_size, void* d_ws, size_t ws_size,
                              hipStream_t stream) {
    (void)in_sizes; (void)n_in; (void)out_size; (void)ws_size;
    const float* x     = (const float*)d_in[0];   // hidden_states [8,2048,1024]
    const float* W     = (const float*)d_in[1];   // W_lin [1024,1024]
    const float* bias  = (const float*)d_in[2];   // b_lin [1024]
    const float* gw    = (const float*)d_in[3];   // gate_w [8,1024]
    const float* left  = (const float*)d_in[4];   // left [8,1024,4]
    const float* right = (const float*)d_in[5];   // right [8,4,1024]
    float* out = (float*)d_out;

    float* coef = (float*)d_ws;                                   // NTOK*8 f32 = 512 KB
    int*   idx  = (int*)((char*)d_ws + (size_t)NTOK * TOPK * RANK * sizeof(float)); // 128 KB

    gate_lora_t_kernel<<<NTOK / 8, 256, 0, stream>>>(x, gw, left, coef, idx);

    dim3 grid(NTOK / BM, OUT / BN);
    lora_moe_gemm_kernel<<<grid, 256, 0, stream>>>(x, W, bias, right, coef, idx, out);
}
